// CausalSelfAttention_82231443849533
// MI455X (gfx1250) — matrix operand order repaired and checked by reading, not compiled, not run on hardware
//
#include <hip/hip_runtime.h>

// ---------------------------------------------------------------------------
// Causal self-attention, B=4 T=2048 D=1024 H=16 HD=64, fp32 in/out.
// bf16 WMMA (v_wmma_f32_16x16x32_bf16) with f32 accumulation everywhere,
// flash-attention inner loop (scores never hit HBM).
// Software-pipelined GEMMs: double-buffered LDS, one barrier per K tile,
// async global->LDS staging (ASYNCcnt) for the bf16 operand of out_gemm.
// ---------------------------------------------------------------------------

typedef __attribute__((ext_vector_type(16))) __bf16 v16bf;
typedef __attribute__((ext_vector_type(8)))  __bf16 v8bf;
typedef __attribute__((ext_vector_type(8)))  float  v8f;

#define BB 4
#define TT 2048
#define DD 1024
#define HH 16
#define HD 64
#define D3 (3 * DD)
#define LP 40   // padded LDS row stride (elements): 80B, 16B-aligned, spreads banks
#define KIT (DD / 32)

// Build a 16-element bf16 A/B fragment from two contiguous 8-element runs.
__device__ __forceinline__ v16bf ld_pair(const __bf16* p0, const __bf16* p1) {
    v8bf lo = *(const v8bf*)p0;
    v8bf hi = *(const v8bf*)p1;
    v16bf r;
#pragma unroll
    for (int i = 0; i < 8; ++i) { r[i] = lo[i]; r[i + 8] = hi[i]; }
    return r;
}

__device__ __forceinline__ v8f wmma_bf16(v16bf a, v16bf b, v8f c) {
    // D = A(16x32 bf16) * B(32x16 bf16) + C(16x16 f32)
    return __builtin_amdgcn_wmma_f32_16x16x32_bf16(
        false, a, false, b, (short)0, c, false, false);
}

// Convert 8 floats (two float4) to a 16-byte bf16 vector.
__device__ __forceinline__ v8bf cvt8(float4 a, float4 b) {
    v8bf r;
    r[0] = (__bf16)a.x; r[1] = (__bf16)a.y; r[2] = (__bf16)a.z; r[3] = (__bf16)a.w;
    r[4] = (__bf16)b.x; r[5] = (__bf16)b.y; r[6] = (__bf16)b.z; r[7] = (__bf16)b.w;
    return r;
}

// ---------------------------------------------------------------------------
// Kernel 1: qkv = x @ w_qkv  (fp32 -> bf16 WMMA), scatter to Q[B,H,T,64]
// (pre-scaled by 1/sqrt(64)), K[B,H,T,64], Vt[B,H,64,T] (transposed).
// Block: 256 thr (8 waves) computes 64x128 output tile. grid=(24,128).
// ---------------------------------------------------------------------------
__global__ __launch_bounds__(256) void qkv_gemm(
    const float* __restrict__ x, const float* __restrict__ w,
    __bf16* __restrict__ Qb, __bf16* __restrict__ Kb, __bf16* __restrict__ Vt) {
    __shared__ __bf16 xs[2][64][LP];    // A tile, row-major (64x32 used)
    __shared__ __bf16 wsT[2][128][LP];  // B tile, transposed (128x32 used)

    const int tid  = threadIdx.x;
    const int lane = tid & 31;
    const int li   = lane & 15;
    const int g    = lane >> 4;
    const int wave = tid >> 5;
    const int wm   = wave >> 1;         // 0..3  (M sub-tile)
    const int wn   = wave & 1;          // 0..1  (N sub-tile)
    const int m0   = blockIdx.y * 64;
    const int n0   = blockIdx.x * 128;

    const int xr = tid >> 2, xc = (tid & 3) * 8;   // x tile: 64 rows x 4 chunks
    const int wr = tid >> 3, wc = (tid & 7) * 16;  // w tile: 32 rows x 8 chunks

    float4 f0, f1, g0, g1, g2, g3;                 // staged global data
    auto load_tiles = [&](int k0) {
        const float* xp = &x[(size_t)(m0 + xr) * DD + k0 + xc];
        f0 = *(const float4*)xp;
        f1 = *(const float4*)(xp + 4);
        const float* wp = &w[(size_t)(k0 + wr) * D3 + n0 + wc];
        g0 = *(const float4*)wp;
        g1 = *(const float4*)(wp + 4);
        g2 = *(const float4*)(wp + 8);
        g3 = *(const float4*)(wp + 12);
    };

    v8f acc[4];
#pragma unroll
    for (int s = 0; s < 4; ++s) acc[s] = 0.f;

    load_tiles(0);
    for (int it = 0; it < KIT; ++it) {
        const int cur = it & 1;
        *(v8bf*)&xs[cur][xr][xc] = cvt8(f0, f1);
        float wv[16] = {g0.x, g0.y, g0.z, g0.w, g1.x, g1.y, g1.z, g1.w,
                        g2.x, g2.y, g2.z, g2.w, g3.x, g3.y, g3.z, g3.w};
#pragma unroll
        for (int j = 0; j < 16; ++j) wsT[cur][wc + j][wr] = (__bf16)wv[j];
        __syncthreads();
        if (it + 1 < KIT) load_tiles((it + 1) * 32);  // overlap with WMMA below

        const int row = wm * 16 + li;
        v16bf a = ld_pair(&xs[cur][row][16 * g], &xs[cur][row][16 * g + 8]);
        v16bf bfr[4];
#pragma unroll
        for (int s = 0; s < 4; ++s) {
            const int col = wn * 64 + s * 16 + li;
            bfr[s] = ld_pair(&wsT[cur][col][16 * g], &wsT[cur][col][16 * g + 8]);
        }
#pragma unroll
        for (int s = 0; s < 4; ++s) acc[s] = wmma_bf16(a, bfr[s], acc[s]);
        // no tail barrier: next iteration writes the other buffer; the next
        // head barrier orders reuse two iterations out.
    }

    // Epilogue: part (q/k/v) is uniform per block -> one block-level branch,
    // three straight-line store loops.
    const int part = n0 >> 10;                 // 0=q 1=k 2=v
    const int nb   = n0 & 1023;
    const int tgb  = m0 + wm * 16 + 8 * g;     // + r
    const int dchb = nb + wn * 64 + li;        // + 16*s
    if (part == 0) {
#pragma unroll
        for (int s = 0; s < 4; ++s)
#pragma unroll
            for (int r = 0; r < 8; ++r) {
                const int tg = tgb + r, dch = dchb + 16 * s;
                const size_t bh = (size_t)((tg >> 11) * HH + (dch >> 6));
                Qb[(bh * TT + (tg & 2047)) * HD + (dch & 63)] =
                    (__bf16)(acc[s][r] * 0.125f);
            }
    } else if (part == 1) {
#pragma unroll
        for (int s = 0; s < 4; ++s)
#pragma unroll
            for (int r = 0; r < 8; ++r) {
                const int tg = tgb + r, dch = dchb + 16 * s;
                const size_t bh = (size_t)((tg >> 11) * HH + (dch >> 6));
                Kb[(bh * TT + (tg & 2047)) * HD + (dch & 63)] = (__bf16)acc[s][r];
            }
    } else {
#pragma unroll
        for (int s = 0; s < 4; ++s)
#pragma unroll
            for (int r = 0; r < 8; ++r) {
                const int tg = tgb + r, dch = dchb + 16 * s;
                const size_t bh = (size_t)((tg >> 11) * HH + (dch >> 6));
                Vt[(bh * HD + (dch & 63)) * TT + (tg & 2047)] = (__bf16)acc[s][r];
            }
    }
}

// ---------------------------------------------------------------------------
// Kernel 2: flash attention. One wave (32 thr) per (head, 16-row q tile).
// grid=(T/16, B*H). S = Q K^T via WMMA, online softmax, O += P V via WMMA.
// ---------------------------------------------------------------------------
__global__ __launch_bounds__(32) void attn_kernel(
    const __bf16* __restrict__ Qb, const __bf16* __restrict__ Kb,
    const __bf16* __restrict__ Vt, __bf16* __restrict__ Ab) {
    __shared__ __bf16 pbuf[16][LP];   // P tile bounce: C-layout -> A-layout

    const int lane = threadIdx.x;
    const int li = lane & 15, g = lane >> 4;
    const int bh = blockIdx.y;
    const int q0 = blockIdx.x * 16;

    // Q fragments (16x32 each), reused across all k chunks. Pre-scaled.
    const size_t baseQ = ((size_t)bh * TT + q0 + li) * HD;
    v16bf qA[2];
#pragma unroll
    for (int f = 0; f < 2; ++f)
        qA[f] = ld_pair(Qb + baseQ + 32 * f + 16 * g,
                        Qb + baseQ + 32 * f + 16 * g + 8);

    v8f o[4];
#pragma unroll
    for (int n = 0; n < 4; ++n) o[n] = 0.f;
    float mrow[8], lrow[8];
#pragma unroll
    for (int r = 0; r < 8; ++r) { mrow[r] = -1e30f; lrow[r] = 0.f; }

    for (int k0 = 0; k0 < q0 + 16; k0 += 32) {
        // Prefetch next chunk of K / V into cache while this chunk computes.
        if (k0 + 32 < q0 + 16) {
            __builtin_prefetch(Kb + ((size_t)bh * TT + k0 + 32 + li) * HD, 0, 0);
            __builtin_prefetch(Vt + ((size_t)bh * HD + li) * TT + k0 + 32, 0, 0);
        }

        // Load all four K fragments, then run the four WMMAs back-to-back.
        v16bf kf[4];
#pragma unroll
        for (int s = 0; s < 2; ++s)
#pragma unroll
            for (int f = 0; f < 2; ++f) {
                const __bf16* kp =
                    Kb + ((size_t)bh * TT + k0 + 16 * s + li) * HD + 32 * f + 16 * g;
                kf[s * 2 + f] = ld_pair(kp, kp + 8);
            }
        v8f s0 = 0.f, s1 = 0.f;
        s0 = wmma_bf16(qA[0], kf[0], s0);
        s0 = wmma_bf16(qA[1], kf[1], s0);
        s1 = wmma_bf16(qA[0], kf[2], s1);
        s1 = wmma_bf16(qA[1], kf[3], s1);

        // Online softmax; a row lives in one 16-lane group at fixed acc idx r.
#pragma unroll
        for (int r = 0; r < 8; ++r) {
            const int q = q0 + r + 8 * g;
            const int k = k0 + li;
            float a0 = (k      > q) ? -1e30f : s0[r];
            float a1 = (k + 16 > q) ? -1e30f : s1[r];
            float mr = fmaxf(a0, a1);
#pragma unroll
            for (int off = 1; off < 16; off <<= 1)
                mr = fmaxf(mr, __shfl_xor(mr, off, 32));
            const float mnew  = fmaxf(mrow[r], mr);
            const float alpha = __expf(mrow[r] - mnew);
            const float p0 = __expf(a0 - mnew);
            const float p1 = __expf(a1 - mnew);
            float rs = p0 + p1;
#pragma unroll
            for (int off = 1; off < 16; off <<= 1)
                rs += __shfl_xor(rs, off, 32);
            lrow[r] = lrow[r] * alpha + rs;
            mrow[r] = mnew;
            o[0][r] *= alpha; o[1][r] *= alpha;
            o[2][r] *= alpha; o[3][r] *= alpha;
            pbuf[r + 8 * g][li]      = (__bf16)p0;
            pbuf[r + 8 * g][li + 16] = (__bf16)p1;
        }
        __syncthreads();

        // O += P(16x32) * V(32x64): load all V fragments, then 4 WMMAs.
        v16bf pA = ld_pair(&pbuf[li][16 * g], &pbuf[li][16 * g + 8]);
        v16bf vf[4];
#pragma unroll
        for (int n = 0; n < 4; ++n) {
            const __bf16* vp =
                Vt + ((size_t)bh * HD + n * 16 + li) * TT + k0 + 16 * g;
            vf[n] = ld_pair(vp, vp + 8);
        }
#pragma unroll
        for (int n = 0; n < 4; ++n) o[n] = wmma_bf16(pA, vf[n], o[n]);
        __syncthreads();
    }

    // Normalize and write attention output as bf16 in [B,T,D] layout.
    const int b = bh / HH, h = bh % HH;
#pragma unroll
    for (int n = 0; n < 4; ++n) {
#pragma unroll
        for (int r = 0; r < 8; ++r) {
            const int t  = q0 + r + 8 * g;
            const int ch = h * HD + n * 16 + li;
            Ab[((size_t)b * TT + t) * DD + ch] = (__bf16)(o[n][r] / lrow[r]);
        }
    }
}

// ---------------------------------------------------------------------------
// Kernel 3: out = attn(bf16) @ w_out(fp32->bf16), fp32 result. grid=(8,128).
// A tile (already bf16 in global) staged via async global->LDS copies with
// ping-pong buffers fenced by s_wait_asynccnt.
// ---------------------------------------------------------------------------
__global__ __launch_bounds__(256) void out_gemm(
    const __bf16* __restrict__ A, const float* __restrict__ w,
    float* __restrict__ out) {
    __shared__ __bf16 xs[2][64][LP];
    __shared__ __bf16 wsT[2][128][LP];

    const int tid  = threadIdx.x;
    const int lane = tid & 31;
    const int li   = lane & 15;
    const int g    = lane >> 4;
    const int wave = tid >> 5;
    const int wm   = wave >> 1;
    const int wn   = wave & 1;
    const int m0   = blockIdx.y * 64;
    const int n0   = blockIdx.x * 128;

    const int xr = tid >> 2, xc = (tid & 3) * 8;   // A tile: 16B per thread
    const int wr = tid >> 3, wc = (tid & 7) * 16;  // w tile
    const unsigned lds_dst[2] = {
        (unsigned)(uintptr_t)&xs[0][xr][xc],
        (unsigned)(uintptr_t)&xs[1][xr][xc]};

    auto issue_async = [&](int k0, int buf) {
        const unsigned goff =
            (unsigned)(((size_t)(m0 + xr) * DD + k0 + xc) * sizeof(__bf16));
        asm volatile("global_load_async_to_lds_b128 %0, %1, %2 offset:0"
                     :: "v"(lds_dst[buf]), "v"(goff), "s"(A) : "memory");
    };

    float4 g0, g1, g2, g3;
    auto load_w = [&](int k0) {
        const float* wp = &w[(size_t)(k0 + wr) * DD + n0 + wc];
        g0 = *(const float4*)wp;
        g1 = *(const float4*)(wp + 4);
        g2 = *(const float4*)(wp + 8);
        g3 = *(const float4*)(wp + 12);
    };

    v8f acc[4];
#pragma unroll
    for (int s = 0; s < 4; ++s) acc[s] = 0.f;

    issue_async(0, 0);
    load_w(0);
    for (int it = 0; it < KIT; ++it) {
        const int cur = it & 1;
        float wv[16] = {g0.x, g0.y, g0.z, g0.w, g1.x, g1.y, g1.z, g1.w,
                        g2.x, g2.y, g2.z, g2.w, g3.x, g3.y, g3.z, g3.w};
#pragma unroll
        for (int j = 0; j < 16; ++j) wsT[cur][wc + j][wr] = (__bf16)wv[j];
        asm volatile("s_wait_asynccnt 0x0" ::: "memory");  // xs[cur] landed
        __syncthreads();
        if (it + 1 < KIT) {              // overlap next tile with this WMMA
            issue_async((it + 1) * 32, cur ^ 1);
            load_w((it + 1) * 32);
        }

        const int row = wm * 16 + li;
        v16bf a = ld_pair(&xs[cur][row][16 * g], &xs[cur][row][16 * g + 8]);
        v16bf bfr[4];
#pragma unroll
        for (int s = 0; s < 4; ++s) {
            const int col = wn * 64 + s * 16 + li;
            bfr[s] = ld_pair(&wsT[cur][col][16 * g], &wsT[cur][col][16 * g + 8]);
        }
#pragma unroll
        for (int s = 0; s < 4; ++s) acc[s] = wmma_bf16(a, bfr[s], acc[s]);
    }

#pragma unroll
    for (int s = 0; s < 4; ++s) {
#pragma unroll
        for (int r = 0; r < 8; ++r) {
            const int tg = m0 + wm * 16 + r + 8 * g;
            const int c  = n0 + wn * 64 + s * 16 + li;
            out[(size_t)tg * DD + c] = acc[s][r];
        }
    }
}

// ---------------------------------------------------------------------------
extern "C" void kernel_launch(void* const* d_in, const int* in_sizes, int n_in,
                              void* d_out, int out_size, void* d_ws, size_t ws_size,
                              hipStream_t stream) {
    (void)in_sizes; (void)n_in; (void)out_size; (void)ws_size;
    const float* x     = (const float*)d_in[0];  // [B,T,D]
    const float* w_qkv = (const float*)d_in[1];  // [D,3D]
    const float* w_out = (const float*)d_in[2];  // [D,D]
    float* out = (float*)d_out;                  // [B,T,D]

    const size_t NQ = (size_t)BB * HH * TT * HD;  // 8,388,608 elems (bf16)
    __bf16* Qb = (__bf16*)d_ws;
    __bf16* Kb = Qb + NQ;
    __bf16* Vt = Kb + NQ;
    __bf16* Ab = Vt + NQ;

    qkv_gemm<<<dim3(D3 / 128, (BB * TT) / 64), 256, 0, stream>>>(x, w_qkv, Qb, Kb, Vt);
    attn_kernel<<<dim3(TT / 16, BB * HH), 32, 0, stream>>>(Qb, Kb, Vt, Ab);
    out_gemm<<<dim3(DD / 128, (BB * TT) / 64), 256, 0, stream>>>(Ab, w_out, out);
}
